// Flux2ParallelSelfAttention_21895743275348
// MI455X (gfx1250) — matrix operand order repaired and checked by reading, not compiled, not run on hardware
//
#include <hip/hip_runtime.h>
#include <hip/hip_bf16.h>
#include <math.h>

typedef __bf16 bf16_t;
typedef __attribute__((ext_vector_type(16))) __bf16 v16bf;
typedef __attribute__((ext_vector_type(8)))  float  v8f;
typedef __attribute__((ext_vector_type(4)))  unsigned u32x4;
typedef __attribute__((ext_vector_type(8)))  unsigned u32x8;

#define S_LEN 4096
#define QDIM  3072
#define HEADS 24
#define HDIM  128
#define NQKV  9216        // 3 * 3072
#define MLPH  12288       // mlp hidden
#define NTOT  33792       // 3*3072 + 2*12288  (W_qkv_mlp row stride)
#define XCOLS 15360       // 3072 + 12288     (concat[attn, gated-mlp])
#define LA_STRIDE 40      // 64B-row +16B stagger, keeps 16B alignment
#define LB_STRIDE 40

__device__ __forceinline__ v8f vzero() {
  v8f z;
#pragma unroll
  for (int e = 0; e < 8; ++e) z[e] = 0.0f;
  return z;
}

__device__ __forceinline__ v8f wmma_bf16(v16bf a, v16bf b, v8f c) {
  return __builtin_amdgcn_wmma_f32_16x16x32_bf16(false, a, false, b, (short)0, c, false, false);
}

// A-matrix fragment (16x32 bf16): lane holds row lane%16.
// elems 0..15 -> K = hi*16 + 0..15 contiguous (matches B fragment)
__device__ __forceinline__ v16bf load_fragA(const bf16_t* row_base, int lane) {
  const bf16_t* p = row_base + ((lane >> 4) & 1) * 16;
  v16bf f;
#pragma unroll
  for (int e = 0; e < 16; ++e) f[e] = p[e];
  return f;
}

// B-matrix fragment (32x16 bf16): lane holds column lane%16,
// K = hi*16 + 0..15 contiguous (tile staged transposed in LDS).
__device__ __forceinline__ v16bf load_fragB(const bf16_t* col_base, int lane) {
  const bf16_t* p = col_base + ((lane >> 4) & 1) * 16;
  v16bf f;
#pragma unroll
  for (int e = 0; e < 16; ++e) f[e] = p[e];
  return f;
}

// CDNA5 async global->LDS copy (ASYNCcnt-tracked), GVS addressing mode.
__device__ __forceinline__ void async_copy_b128(unsigned lds_off,
                                                const bf16_t* sbase,
                                                unsigned byte_off) {
  asm volatile("global_load_async_to_lds_b128 %0, %1, %2"
               :: "v"(lds_off), "v"(byte_off), "s"(sbase)
               : "memory");
}
__device__ __forceinline__ void wait_async0() {
  asm volatile("s_wait_asynccnt 0" ::: "memory");
}

// ---------------------------------------------------------------------------
// Kernel 1: QKV GEMM (64x128 tile = one head) + fused RMSNorm + RoPE epilogue
// grid: (S/64, 72)  [24 q-heads | 24 k-heads | 24 v-heads], 256 threads
// ---------------------------------------------------------------------------
__global__ __launch_bounds__(256)
void qkv_gemm_kernel(const float* __restrict__ hid, const float* __restrict__ W,
                     const float* __restrict__ cosv, const float* __restrict__ sinv,
                     const float* __restrict__ nqw, const float* __restrict__ nkw,
                     bf16_t* __restrict__ Qb, bf16_t* __restrict__ Kb,
                     bf16_t* __restrict__ Vb) {
  __shared__ __align__(16) bf16_t la[64 * LA_STRIDE];
  __shared__ __align__(16) bf16_t lb[128 * LB_STRIDE];
  __shared__ float lc[64 * 128];

  const int tid  = threadIdx.x;
  const int lane = tid & 31;
  const int w    = tid >> 5;
  const int wm   = w >> 2;          // 0..1
  const int wn   = w & 3;           // 0..3
  const int s0   = blockIdx.x * 64;
  const int ht   = blockIdx.y;      // 0..71
  const int n0   = ht * 128;

  v8f acc[2][2];
#pragma unroll
  for (int i = 0; i < 2; ++i)
#pragma unroll
    for (int j = 0; j < 2; ++j) acc[i][j] = vzero();

  const int ar = tid >> 2;          // 0..63 : A row
  const int ac = (tid & 3) * 8;     // A col group (8 floats)
  const int bk = tid >> 3;          // 0..31 : B k-row
  const int bc = (tid & 7) * 16;    // B col group (16 floats)

  for (int k0 = 0; k0 < QDIM; k0 += 32) {
    __syncthreads();
    { // A tile 64x32 fp32 -> bf16 LDS (row-major)
      const float* src = hid + (size_t)(s0 + ar) * QDIM + k0 + ac;
      float4 f0 = *(const float4*)(src);
      float4 f1 = *(const float4*)(src + 4);
      bf16_t* dst = la + ar * LA_STRIDE + ac;
      dst[0] = (bf16_t)f0.x; dst[1] = (bf16_t)f0.y;
      dst[2] = (bf16_t)f0.z; dst[3] = (bf16_t)f0.w;
      dst[4] = (bf16_t)f1.x; dst[5] = (bf16_t)f1.y;
      dst[6] = (bf16_t)f1.z; dst[7] = (bf16_t)f1.w;
    }
    { // B tile 32x128 fp32 -> bf16 LDS, stored transposed lb[n][k]
      const float* src = W + (size_t)(k0 + bk) * NTOT + n0 + bc;
      __builtin_prefetch(src + (size_t)32 * NTOT, 0, 1);  // next K stage
#pragma unroll
      for (int i = 0; i < 16; i += 4) {
        float4 f = *(const float4*)(src + i);
        lb[(bc + i + 0) * LB_STRIDE + bk] = (bf16_t)f.x;
        lb[(bc + i + 1) * LB_STRIDE + bk] = (bf16_t)f.y;
        lb[(bc + i + 2) * LB_STRIDE + bk] = (bf16_t)f.z;
        lb[(bc + i + 3) * LB_STRIDE + bk] = (bf16_t)f.w;
      }
    }
    __syncthreads();
    v16bf af[2], bfr[2];
#pragma unroll
    for (int i = 0; i < 2; ++i)
      af[i] = load_fragA(la + (wm * 32 + i * 16 + (lane & 15)) * LA_STRIDE, lane);
#pragma unroll
    for (int j = 0; j < 2; ++j)
      bfr[j] = load_fragB(lb + (wn * 32 + j * 16 + (lane & 15)) * LB_STRIDE, lane);
#pragma unroll
    for (int i = 0; i < 2; ++i)
#pragma unroll
      for (int j = 0; j < 2; ++j)
        acc[i][j] = wmma_bf16(af[i], bfr[j], acc[i][j]);
  }

  // spill fp32 tile to LDS for cross-wave per-row RMSNorm
  const int hi  = (lane >> 4) & 1;
  const int c16 = lane & 15;
#pragma unroll
  for (int i = 0; i < 2; ++i)
#pragma unroll
    for (int j = 0; j < 2; ++j)
#pragma unroll
      for (int r = 0; r < 8; ++r)
        lc[(wm * 32 + i * 16 + hi * 8 + r) * 128 + (wn * 32 + j * 16 + c16)] =
            acc[i][j][r];
  __syncthreads();

  // 4 threads per row, 32 cols each
  const int row = tid >> 2;
  const int q   = tid & 3;
  const int s   = s0 + row;
  float x[32];
  float ss = 0.0f;
#pragma unroll
  for (int i = 0; i < 32; ++i) {
    x[i] = lc[row * 128 + q * 32 + i];
    ss += x[i] * x[i];
  }
  ss += __shfl_xor(ss, 1, 32);
  ss += __shfl_xor(ss, 2, 32);

  const int kind = ht / 24;   // 0=q 1=k 2=v
  const int head = ht % 24;
  bf16_t* dstbase;
  if (kind == 2) {
    dstbase = Vb;
  } else {
    float rinv = rsqrtf(ss * (1.0f / 128.0f) + 1e-5f);
    const float* nw = (kind == 0) ? nqw : nkw;
#pragma unroll
    for (int i = 0; i < 32; ++i) x[i] = x[i] * rinv * nw[q * 32 + i];
#pragma unroll
    for (int p = 0; p < 16; ++p) {      // interleaved RoPE
      int fi = q * 16 + p;
      float c  = cosv[s * 64 + fi];
      float sn = sinv[s * 64 + fi];
      float x1 = x[2 * p], x2 = x[2 * p + 1];
      x[2 * p]     = x1 * c - x2 * sn;
      x[2 * p + 1] = x2 * c + x1 * sn;
    }
    dstbase = (kind == 0) ? Qb : Kb;
  }
  bf16_t* dst = dstbase + (size_t)head * S_LEN * HDIM + (size_t)s * HDIM + q * 32;
#pragma unroll
  for (int i = 0; i < 32; ++i) dst[i] = (bf16_t)x[i];
}

// ---------------------------------------------------------------------------
// Kernel 2: MLP GEMM — two K-passes (gate tile n, value tile n+12288),
// fused SiLU gate, bf16 out into concat-X columns [3072, 15360)
// grid: (S/64, 96), 256 threads
// ---------------------------------------------------------------------------
__global__ __launch_bounds__(256)
void mlp_gemm_kernel(const float* __restrict__ hid, const float* __restrict__ W,
                     bf16_t* __restrict__ Xb) {
  __shared__ __align__(16) bf16_t la[64 * LA_STRIDE];
  __shared__ __align__(16) bf16_t lb[128 * LB_STRIDE];

  const int tid  = threadIdx.x;
  const int lane = tid & 31;
  const int w    = tid >> 5;
  const int wm   = w >> 2;
  const int wn   = w & 3;
  const int s0   = blockIdx.x * 64;
  const int j0   = blockIdx.y * 128;

  v8f acc[2][2][2];   // [pass][i][j]
#pragma unroll
  for (int g = 0; g < 2; ++g)
#pragma unroll
    for (int i = 0; i < 2; ++i)
#pragma unroll
      for (int j = 0; j < 2; ++j) acc[g][i][j] = vzero();

  const int ar = tid >> 2;
  const int ac = (tid & 3) * 8;
  const int bk = tid >> 3;
  const int bc = (tid & 7) * 16;

#pragma unroll
  for (int g = 0; g < 2; ++g) {
    const float* Wg = W + NQKV + (size_t)g * MLPH + j0;
    for (int k0 = 0; k0 < QDIM; k0 += 32) {
      __syncthreads();
      {
        const float* src = hid + (size_t)(s0 + ar) * QDIM + k0 + ac;
        float4 f0 = *(const float4*)(src);
        float4 f1 = *(const float4*)(src + 4);
        bf16_t* dst = la + ar * LA_STRIDE + ac;
        dst[0] = (bf16_t)f0.x; dst[1] = (bf16_t)f0.y;
        dst[2] = (bf16_t)f0.z; dst[3] = (bf16_t)f0.w;
        dst[4] = (bf16_t)f1.x; dst[5] = (bf16_t)f1.y;
        dst[6] = (bf16_t)f1.z; dst[7] = (bf16_t)f1.w;
      }
      {
        const float* src = Wg + (size_t)(k0 + bk) * NTOT + bc;
        __builtin_prefetch(src + (size_t)32 * NTOT, 0, 1);
#pragma unroll
        for (int i = 0; i < 16; i += 4) {
          float4 f = *(const float4*)(src + i);
          lb[(bc + i + 0) * LB_STRIDE + bk] = (bf16_t)f.x;
          lb[(bc + i + 1) * LB_STRIDE + bk] = (bf16_t)f.y;
          lb[(bc + i + 2) * LB_STRIDE + bk] = (bf16_t)f.z;
          lb[(bc + i + 3) * LB_STRIDE + bk] = (bf16_t)f.w;
        }
      }
      __syncthreads();
      v16bf af[2], bfr[2];
#pragma unroll
      for (int i = 0; i < 2; ++i)
        af[i] = load_fragA(la + (wm * 32 + i * 16 + (lane & 15)) * LA_STRIDE, lane);
#pragma unroll
      for (int j = 0; j < 2; ++j)
        bfr[j] = load_fragB(lb + (wn * 32 + j * 16 + (lane & 15)) * LB_STRIDE, lane);
#pragma unroll
      for (int i = 0; i < 2; ++i)
#pragma unroll
        for (int j = 0; j < 2; ++j)
          acc[g][i][j] = wmma_bf16(af[i], bfr[j], acc[g][i][j]);
    }
  }

  const int hi  = (lane >> 4) & 1;
  const int c16 = lane & 15;
#pragma unroll
  for (int i = 0; i < 2; ++i)
#pragma unroll
    for (int j = 0; j < 2; ++j)
#pragma unroll
      for (int r = 0; r < 8; ++r) {
        int row = wm * 32 + i * 16 + hi * 8 + r;
        int col = wn * 32 + j * 16 + c16;
        float m1 = acc[0][i][j][r];
        float m2 = acc[1][i][j][r];
        float sl = m1 / (1.0f + __expf(-m1));     // SiLU
        Xb[(size_t)(s0 + row) * XCOLS + QDIM + j0 + col] = (bf16_t)(sl * m2);
      }
}

// ---------------------------------------------------------------------------
// Kernel 3: flash attention per head. 64 queries/block, 4 waves (16 q each),
// 32 keys/iteration, online softmax fp32, WMMA for QK^T and PV.
// Q/K tiles move via CDNA5 async global->LDS DMA (ASYNCcnt).
// grid: (24, S/64), 128 threads
// ---------------------------------------------------------------------------
__global__ __launch_bounds__(128)
void attn_kernel(const bf16_t* __restrict__ Qb, const bf16_t* __restrict__ Kb,
                 const bf16_t* __restrict__ Vb, bf16_t* __restrict__ Xb) {
  __shared__ __align__(16) bf16_t lq[64 * HDIM];
  __shared__ __align__(16) bf16_t lk[32 * HDIM];
  __shared__ __align__(16) bf16_t lvt[HDIM * 32];   // V transposed: lvt[d][key]
  __shared__ __align__(16) bf16_t pw[4 * 16 * 32];  // per-wave P scratch

  const int tid  = threadIdx.x;
  const int lane = tid & 31;
  const int w    = tid >> 5;
  const int head = blockIdx.x;
  const int q0   = blockIdx.y * 64;
  const size_t hbase = (size_t)head * S_LEN * HDIM;

  { // Q tile 64x128 bf16 via async DMA: 128 thr x 128B
    int r = tid >> 1, c0 = (tid & 1) * 64;
    unsigned ldso = (unsigned)(uintptr_t)lq + (unsigned)(r * HDIM + c0) * 2u;
    unsigned goff = (unsigned)(((q0 + r) * HDIM + c0) * 2);
    const bf16_t* qbase = Qb + hbase;
#pragma unroll
    for (int i = 0; i < 8; ++i)
      async_copy_b128(ldso + i * 16, qbase, goff + i * 16);
  }
  wait_async0();
  __syncthreads();

  v16bf qf[4];
#pragma unroll
  for (int kk = 0; kk < 4; ++kk)
    qf[kk] = load_fragA(lq + (w * 16 + (lane & 15)) * HDIM + kk * 32, lane);

  v8f ao[8];
#pragma unroll
  for (int c = 0; c < 8; ++c) ao[c] = vzero();
  float mrow[8], lrow[8];
#pragma unroll
  for (int r = 0; r < 8; ++r) { mrow[r] = -1e30f; lrow[r] = 0.0f; }

  const float scale = 0.08838834764831845f;  // 1/sqrt(128)
  const int hi = (lane >> 4) & 1;

  for (int j0 = 0; j0 < S_LEN; j0 += 32) {
    __syncthreads();
    { // K block 32x128 via async DMA: 128 thr x 64B
      int r = tid >> 2, c0 = (tid & 3) * 32;
      unsigned ldso = (unsigned)(uintptr_t)lk + (unsigned)(r * HDIM + c0) * 2u;
      unsigned goff = (unsigned)(((j0 + r) * HDIM + c0) * 2);
      const bf16_t* kbase = Kb + hbase;
#pragma unroll
      for (int i = 0; i < 4; ++i)
        async_copy_b128(ldso + i * 16, kbase, goff + i * 16);
    }
    { // V block transposed into lvt[d][key] (needs element transpose -> VGPR path)
      int r = tid >> 2, c0 = (tid & 3) * 32;
      const bf16_t* src = Vb + hbase + (size_t)(j0 + r) * HDIM + c0;
#pragma unroll
      for (int i = 0; i < 32; ++i) lvt[(c0 + i) * 32 + r] = src[i];
    }
    wait_async0();
    __syncthreads();

    // scores S = Q @ K^T for 32 keys (two 16-wide tiles)
    v8f sa0 = vzero(), sa1 = vzero();
#pragma unroll
    for (int kk = 0; kk < 4; ++kk) {
      v16bf b0 = load_fragB(lk + (lane & 15) * HDIM + kk * 32, lane);
      v16bf b1 = load_fragB(lk + (16 + (lane & 15)) * HDIM + kk * 32, lane);
      sa0 = wmma_bf16(qf[kk], b0, sa0);
      sa1 = wmma_bf16(qf[kk], b1, sa1);
    }

    // online softmax (row = r + hi*8, replicated across 16-lane half)
    float pr0[8], pr1[8], alpha[8];
#pragma unroll
    for (int r = 0; r < 8; ++r) {
      float v0 = sa0[r] * scale, v1 = sa1[r] * scale;
      float mx = fmaxf(v0, v1);
      mx = fmaxf(mx, __shfl_xor(mx, 1, 32));
      mx = fmaxf(mx, __shfl_xor(mx, 2, 32));
      mx = fmaxf(mx, __shfl_xor(mx, 4, 32));
      mx = fmaxf(mx, __shfl_xor(mx, 8, 32));
      float mnew = fmaxf(mrow[r], mx);
      float a  = __expf(mrow[r] - mnew);
      float p0 = __expf(v0 - mnew);
      float p1 = __expf(v1 - mnew);
      float ps = p0 + p1;
      ps += __shfl_xor(ps, 1, 32);
      ps += __shfl_xor(ps, 2, 32);
      ps += __shfl_xor(ps, 4, 32);
      ps += __shfl_xor(ps, 8, 32);
      lrow[r] = lrow[r] * a + ps;
      mrow[r] = mnew;
      alpha[r] = a;
      pr0[r] = p0; pr1[r] = p1;
    }
#pragma unroll
    for (int c = 0; c < 8; ++c)
#pragma unroll
      for (int r = 0; r < 8; ++r) ao[c][r] *= alpha[r];

    // transpose P (C-layout -> A-fragment layout) via per-wave LDS scratch
    bf16_t* pws = pw + w * (16 * 32);
#pragma unroll
    for (int r = 0; r < 8; ++r) {
      pws[(hi * 8 + r) * 32 + (lane & 15)]      = (bf16_t)pr0[r];
      pws[(hi * 8 + r) * 32 + 16 + (lane & 15)] = (bf16_t)pr1[r];
    }
    v16bf pf = load_fragA(pws + (lane & 15) * 32, lane);
#pragma unroll
    for (int c = 0; c < 8; ++c) {
      v16bf vb = load_fragB(lvt + (c * 16 + (lane & 15)) * 32, lane);
      ao[c] = wmma_bf16(pf, vb, ao[c]);
    }
  }

  // normalize + write attn output into concat-X columns [0, 3072)
#pragma unroll
  for (int r = 0; r < 8; ++r) {
    float inv = (lrow[r] > 0.0f) ? 1.0f / lrow[r] : 0.0f;
    int s = q0 + w * 16 + hi * 8 + r;
    bf16_t* dst = Xb + (size_t)s * XCOLS + head * HDIM;
#pragma unroll
    for (int c = 0; c < 8; ++c)
      dst[c * 16 + (lane & 15)] = (bf16_t)(ao[c][r] * inv);
  }
}

// ---------------------------------------------------------------------------
// Kernel 4: out = X(4096x15360 bf16) @ W_out(fp32->bf16) + b_out, fp32 out.
// The bf16 A tile moves via TDM tensor_load_to_lds (TENSORcnt), with the D#
// pad fields reproducing the stride-40 padded LDS layout directly.
// grid: (S/64, 24), 256 threads
// ---------------------------------------------------------------------------
__global__ __launch_bounds__(256)
void out_gemm_kernel(const bf16_t* __restrict__ X, const float* __restrict__ Wo,
                     const float* __restrict__ bias, float* __restrict__ out) {
  __shared__ __align__(16) bf16_t la[64 * LA_STRIDE];
  __shared__ __align__(16) bf16_t lb[128 * LB_STRIDE];

  const int tid  = threadIdx.x;
  const int lane = tid & 31;
  const int w    = tid >> 5;
  const int wm   = w >> 2;
  const int wn   = w & 3;
  const int s0   = blockIdx.x * 64;
  const int n0   = blockIdx.y * 128;

  v8f acc[2][2];
#pragma unroll
  for (int i = 0; i < 2; ++i)
#pragma unroll
    for (int j = 0; j < 2; ++j) acc[i][j] = vzero();

  const int bk = tid >> 3;
  const int bc = (tid & 7) * 16;

  // D# group 1 (k0-invariant): 2D tile 32(k) x 64(rows), 2B elems,
  // pad 4 DWORDs after every 16 DWORDs -> LDS row stride 80B (= LA_STRIDE).
  u32x8 g1;
  g1[0] = (1u << 16) | (1u << 20) | (3u << 22) | (3u << 25);
  g1[1] = ((unsigned)XCOLS & 0xFFFFu) << 16;          // tensor_dim0 lo16
  g1[2] = (((unsigned)XCOLS >> 16) & 0xFFFFu) | (((unsigned)S_LEN & 0xFFFFu) << 16);
  g1[3] = (((unsigned)S_LEN >> 16) & 0xFFFFu) | (32u << 16);   // tile_dim0 = 32
  g1[4] = 64u;                                        // tile_dim1 = 64, tile_dim2 = 0
  g1[5] = (unsigned)XCOLS;                            // tensor_dim0_stride lo32
  g1[6] = 0u;
  g1[7] = 0u;
  const unsigned lds_a = (unsigned)(uintptr_t)la;

  for (int k0 = 0; k0 < XCOLS; k0 += 32) {
    __syncthreads();
    if (w == 0) { // one wave issues the TDM DMA for the A tile (EXEC-agnostic op)
      unsigned long long ga = (unsigned long long)(uintptr_t)X +
          ((unsigned long long)s0 * XCOLS + (unsigned)k0) * 2ull;
      u32x4 g0;
      g0[0] = 1u;                                     // count = 1
      g0[1] = lds_a;                                  // LDS dest
      g0[2] = (unsigned)ga;                           // global_addr lo
      g0[3] = ((unsigned)(ga >> 32) & 0x01FFFFFFu) | 0x80000000u;  // hi + type=2
      asm volatile("tensor_load_to_lds %0, %1" :: "s"(g0), "s"(g1) : "memory");
      __builtin_amdgcn_s_wait_tensorcnt(0);
    }
    { // W_out tile fp32 -> bf16 transposed (convert path: DMA cannot narrow)
      const float* src = Wo + (size_t)(k0 + bk) * QDIM + n0 + bc;
      __builtin_prefetch(src + (size_t)32 * QDIM, 0, 1);
#pragma unroll
      for (int i = 0; i < 16; i += 4) {
        float4 f = *(const float4*)(src + i);
        lb[(bc + i + 0) * LB_STRIDE + bk] = (bf16_t)f.x;
        lb[(bc + i + 1) * LB_STRIDE + bk] = (bf16_t)f.y;
        lb[(bc + i + 2) * LB_STRIDE + bk] = (bf16_t)f.z;
        lb[(bc + i + 3) * LB_STRIDE + bk] = (bf16_t)f.w;
      }
    }
    __syncthreads();
    v16bf af[2], bfr[2];
#pragma unroll
    for (int i = 0; i < 2; ++i)
      af[i] = load_fragA(la + (wm * 32 + i * 16 + (lane & 15)) * LA_STRIDE, lane);
#pragma unroll
    for (int j = 0; j < 2; ++j)
      bfr[j] = load_fragB(lb + (wn * 32 + j * 16 + (lane & 15)) * LB_STRIDE, lane);
#pragma unroll
    for (int i = 0; i < 2; ++i)
#pragma unroll
      for (int j = 0; j < 2; ++j)
        acc[i][j] = wmma_bf16(af[i], bfr[j], acc[i][j]);
  }

  const int hi  = (lane >> 4) & 1;
  const int c16 = lane & 15;
#pragma unroll
  for (int i = 0; i < 2; ++i)
#pragma unroll
    for (int j = 0; j < 2; ++j)
#pragma unroll
      for (int r = 0; r < 8; ++r) {
        int row = wm * 32 + i * 16 + hi * 8 + r;
        int col = wn * 32 + j * 16 + c16;
        out[(size_t)(s0 + row) * QDIM + n0 + col] = acc[i][j][r] + bias[n0 + col];
      }
}

// ---------------------------------------------------------------------------
extern "C" void kernel_launch(void* const* d_in, const int* in_sizes, int n_in,
                              void* d_out, int out_size, void* d_ws, size_t ws_size,
                              hipStream_t stream) {
  const float* hid  = (const float*)d_in[0];
  const float* cosv = (const float*)d_in[1];
  const float* sinv = (const float*)d_in[2];
  const float* Wqm  = (const float*)d_in[3];
  const float* Wout = (const float*)d_in[4];
  const float* bout = (const float*)d_in[5];
  const float* nqw  = (const float*)d_in[6];
  const float* nkw  = (const float*)d_in[7];

  // workspace layout (bf16): Q | K | V | X(concat attn+mlp)  ~192 MB total
  bf16_t* Qb = (bf16_t*)d_ws;
  bf16_t* Kb = Qb + (size_t)HEADS * S_LEN * HDIM;
  bf16_t* Vb = Kb + (size_t)HEADS * S_LEN * HDIM;
  bf16_t* Xb = Vb + (size_t)HEADS * S_LEN * HDIM;

  qkv_gemm_kernel<<<dim3(S_LEN / 64, 72), 256, 0, stream>>>(
      hid, Wqm, cosv, sinv, nqw, nkw, Qb, Kb, Vb);
  mlp_gemm_kernel<<<dim3(S_LEN / 64, MLPH / 128), 256, 0, stream>>>(hid, Wqm, Xb);
  attn_kernel<<<dim3(HEADS, S_LEN / 64), 128, 0, stream>>>(Qb, Kb, Vb, Xb);
  out_gemm_kernel<<<dim3(S_LEN / 64, QDIM / 128), 256, 0, stream>>>(
      Xb, Wout, bout, (float*)d_out);
}
